// attention_17437567222073
// MI455X (gfx1250) — matrix operand order repaired and checked by reading, not compiled, not run on hardware
//
#include <hip/hip_runtime.h>
#include <math.h>

#define HH 300
#define BB 32
#define SS 50
#define TT 50
#define VV 19495
#define KP 608            // 2H=600 padded to multiple of 32 for WMMA K
#define MM (BB * TT)      // 1600 rows of the final GEMM
#define MP 1664           // 1600 padded to 13*8*16
#define VP 19520          // 19495 padded to multiple of 64 (wave N-tile)

typedef __attribute__((ext_vector_type(16))) __bf16 v16bf;
typedef __attribute__((ext_vector_type(8)))  float  v8f;

__device__ __forceinline__ float sigm(float x) { return 1.f / (1.f + expf(-x)); }

__device__ __forceinline__ unsigned short f2bf(float f) {
  unsigned int u = __float_as_uint(f);
  unsigned int r = u + 0x7fffu + ((u >> 16) & 1u);   // round-to-nearest-even
  return (unsigned short)(r >> 16);
}

// 300-length dot product, float4 vectorized (w: global row 1200B-aligned, x: LDS 16B-aligned)
__device__ __forceinline__ float dot300(const float* __restrict__ w, const float* __restrict__ x) {
  const float4* w4 = (const float4*)w;
  const float4* x4 = (const float4*)x;
  float acc = 0.f;
#pragma unroll 5
  for (int q = 0; q < 75; ++q) {
    float4 a = w4[q], b = x4[q];
    acc += a.x * b.x + a.y * b.y + a.z * b.z + a.w * b.w;
  }
  return acc;
}

// ---------------- utility kernels ----------------
__global__ void zero_u16(unsigned short* p, int n) {
  int i = blockIdx.x * blockDim.x + threadIdx.x;
  if (i < n) p[i] = 0;
}
// out_W [V,600] f32 -> [VP,KP] bf16, zero padded
__global__ void convert_w_bf16(const float* __restrict__ W, unsigned short* __restrict__ Wb) {
  int i = blockIdx.x * blockDim.x + threadIdx.x;
  if (i >= VP * KP) return;
  int v = i / KP, k = i % KP;
  float x = (v < VV && k < 2 * HH) ? W[v * 2 * HH + k] : 0.f;
  Wb[i] = f2bf(x);
}
// embedding gather: out[tok,:] = emb[ids[tok],:]
__global__ void embed_kernel(const int* __restrict__ ids, const float* __restrict__ emb,
                             float* __restrict__ out, int n_tok) {
  int i = blockIdx.x * blockDim.x + threadIdx.x;
  if (i >= n_tok * HH) return;
  int tok = i / HH, j = i % HH;
  out[i] = emb[(size_t)ids[tok] * HH + j];
}

// ---------------- f32 tiled SGEMM: C[M,N] = A[M,K] @ W[N,K]^T + bias[N] ----------------
__global__ void sgemm_nt_bias(const float* __restrict__ A, const float* __restrict__ W,
                              const float* __restrict__ bias, float* __restrict__ C,
                              int M, int N, int K) {
  __shared__ float As[16][16];
  __shared__ float Ws[16][17];
  int tx = threadIdx.x, ty = threadIdx.y;
  int row = blockIdx.y * 16 + ty;   // M index
  int col = blockIdx.x * 16 + tx;   // N index
  float acc = 0.f;
  for (int k0 = 0; k0 < K; k0 += 16) {
    int ka = k0 + tx;
    As[ty][tx] = (row < M && ka < K) ? A[(size_t)row * K + ka] : 0.f;
    int wr = blockIdx.x * 16 + ty;  // N index for the W tile
    Ws[ty][tx] = (wr < N && ka < K) ? W[(size_t)wr * K + ka] : 0.f;
    __syncthreads();
#pragma unroll
    for (int kk = 0; kk < 16; ++kk) acc += As[ty][kk] * Ws[tx][kk];
    __syncthreads();
  }
  if (row < M && col < N) C[(size_t)row * N + col] = acc + bias[col];
}

// ---------------- fused encoder: one block per batch row, all 50 steps in-kernel ----------------
__global__ void __launch_bounds__(256) encoder_kernel(
    const float* __restrict__ xp, const float* __restrict__ Whh, const float* __restrict__ bhh,
    float* __restrict__ enc_out, float* __restrict__ hfin) {
  __shared__ __align__(16) float h_s[HH];
  __shared__ float gh_s[3 * HH];
  int b = blockIdx.x;
  for (int j = threadIdx.x; j < HH; j += blockDim.x) h_s[j] = 0.f;   // h0 = zeros
  __syncthreads();
  for (int s = 0; s < SS; ++s) {
    // gh = h @ Whh^T + bhh
    for (int j = threadIdx.x; j < 3 * HH; j += blockDim.x)
      gh_s[j] = bhh[j] + dot300(Whh + (size_t)j * HH, h_s);
    __syncthreads();
    // gates + state update (each thread touches only its own h_s[j])
    const float* xrow = xp + (size_t)(b * SS + s) * 3 * HH;
    for (int j = threadIdx.x; j < HH; j += blockDim.x) {
      float r = sigm(xrow[j] + gh_s[j]);
      float z = sigm(xrow[HH + j] + gh_s[HH + j]);
      float n = tanhf(xrow[2 * HH + j] + r * gh_s[2 * HH + j]);
      float hn = (1.f - z) * n + z * h_s[j];
      h_s[j] = hn;
      enc_out[(size_t)(b * SS + s) * HH + j] = hn;
    }
    __syncthreads();
  }
  for (int j = threadIdx.x; j < HH; j += blockDim.x) hfin[b * HH + j] = h_s[j];
}

// ---------------- fused decoder: GRU cell + attention, all 50 steps in-kernel ----------------
__global__ void __launch_bounds__(256) decoder_kernel(
    const float* __restrict__ xp, const float* __restrict__ Whh, const float* __restrict__ bhh,
    const float* __restrict__ Wdec, const float* __restrict__ bdec,
    const float* __restrict__ Wv, const float* __restrict__ bv,
    const float* __restrict__ ge, const float* __restrict__ enc_out,
    const float* __restrict__ h0, unsigned short* __restrict__ Abf) {
  __shared__ __align__(16) float h_s[HH];      // current hidden (= hn after gate phase)
  __shared__ __align__(16) float gd_s[HH];
  __shared__ float gh_s[3 * HH];
  __shared__ float w_s[SS];
  __shared__ float red0;
  int b = blockIdx.x;
  int tid = threadIdx.x;
  int lane = tid & 31, warp = tid >> 5;
  const float* gerow0 = ge + (size_t)b * SS * HH;
  const float* eorow0 = enc_out + (size_t)b * SS * HH;
  for (int j = tid; j < HH; j += blockDim.x) h_s[j] = h0[b * HH + j];
  __syncthreads();
  for (int t = 0; t < TT; ++t) {
    // gh = h @ Whh^T + bhh
    for (int j = tid; j < 3 * HH; j += blockDim.x)
      gh_s[j] = bhh[j] + dot300(Whh + (size_t)j * HH, h_s);
    __syncthreads();
    // GRU cell -> hn, bank into A-matrix cols 300..599 (bf16)
    const float* xrow = xp + (size_t)(b * TT + t) * 3 * HH;
    size_t arow = (size_t)(b * TT + t) * KP;
    for (int j = tid; j < HH; j += blockDim.x) {
      float r = sigm(xrow[j] + gh_s[j]);
      float z = sigm(xrow[HH + j] + gh_s[HH + j]);
      float n = tanhf(xrow[2 * HH + j] + r * gh_s[2 * HH + j]);
      float hn = (1.f - z) * n + z * h_s[j];
      h_s[j] = hn;
      Abf[arow + HH + j] = f2bf(hn);
    }
    __syncthreads();
    // gd = hn @ Wdec^T + bdec
    for (int j = tid; j < HH; j += blockDim.x)
      gd_s[j] = bdec[j] + dot300(Wdec + (size_t)j * HH, h_s);
    __syncthreads();
    // scores: each wave handles ~6-7 score rows, 32 lanes split the 300-elem tanh reduction
    for (int s = warp; s < SS; s += 8) {
      const float* gerow = gerow0 + (size_t)s * HH;
      float acc = 0.f;
      for (int k = lane; k < HH; k += 32)
        acc += tanhf(gerow[k] + gd_s[k]) * Wv[k];
#pragma unroll
      for (int o = 16; o > 0; o >>= 1) acc += __shfl_down(acc, o);
      if (lane == 0) w_s[s] = acc + bv[0];
    }
    __syncthreads();
    if (tid == 0) {   // softmax over 50 (trivial serial)
      float mx = w_s[0];
      for (int s = 1; s < SS; ++s) mx = fmaxf(mx, w_s[s]);
      float sum = 0.f;
      for (int s = 0; s < SS; ++s) { float e = expf(w_s[s] - mx); w_s[s] = e; sum += e; }
      red0 = 1.f / sum;
    }
    __syncthreads();
    // ctx = sum_s w[s] * enc_out[b,s,:] -> A-matrix cols 0..299 (bf16)
    float inv = red0;
    for (int j = tid; j < HH; j += blockDim.x) {
      float acc = 0.f;
      for (int s = 0; s < SS; ++s) acc += w_s[s] * eorow0[(size_t)s * HH + j];
      Abf[arow + j] = f2bf(acc * inv);
    }
    __syncthreads();
  }
}

// ---------------- final vocab projection: bf16 WMMA GEMM ----------------
// C[r,v] = sum_k A[r,k]*Wb[v,k] + bias[v]; A:[MP,KP] bf16, Wb:[VP,KP] bf16
__global__ void __launch_bounds__(256) wmma_out_gemm(
    const unsigned short* __restrict__ A, const unsigned short* __restrict__ Wb,
    const float* __restrict__ bias, float* __restrict__ out) {
  int lane = threadIdx.x & 31;
  int wave = threadIdx.x >> 5;
  int ln = lane & 15;       // M-row within A-frag / N-col within B,C frags
  int hi = lane >> 4;
  int mtile = (blockIdx.y * 8 + wave) * 16;
  int ntile = blockIdx.x * 64;

  v8f c[4];
  v8f zf = {0.f, 0.f, 0.f, 0.f, 0.f, 0.f, 0.f, 0.f};
  c[0] = zf; c[1] = zf; c[2] = zf; c[3] = zf;

  const unsigned short* arow = A + (size_t)(mtile + ln) * KP;
  for (int k0 = 0; k0 < KP; k0 += 32) {
    // A-frag (ISA 16-bit 16x32 layout): lane<16 -> K {k0..k0+7, k0+16..k0+23}; lane>=16 -> +8
    union { v16bf v; uint4 q[2]; } a;
    a.q[0] = *(const uint4*)(arow + k0 + 8 * hi);
    a.q[1] = *(const uint4*)(arow + k0 + 16 + 8 * hi);
#pragma unroll
    for (int nt = 0; nt < 4; ++nt) {
      // B-frag: lanes 0-15 hold K {k0..k0+7, k0+16..k0+23} of col n; lanes 16-31 -> +8
      const unsigned short* brow = Wb + (size_t)(ntile + nt * 16 + ln) * KP + k0 + 8 * hi;
      union { v16bf v; uint4 q[2]; } bf;
      bf.q[0] = *(const uint4*)(brow);
      bf.q[1] = *(const uint4*)(brow + 16);
      c[nt] = __builtin_amdgcn_wmma_f32_16x16x32_bf16(
          false, a.v, false, bf.v, (short)0, c[nt], false, false);
    }
  }
#pragma unroll
  for (int nt = 0; nt < 4; ++nt) {
    int col = ntile + nt * 16 + ln;
    if (col < VV) {
      float bb = bias[col];
      union { v8f v; float f[8]; } cu; cu.v = c[nt];
#pragma unroll
      for (int j = 0; j < 8; ++j) {
        int r = mtile + j + 8 * hi;
        if (r < MM) out[(size_t)r * VV + col] = cu.f[j] + bb;
      }
    }
  }
}

extern "C" void kernel_launch(void* const* d_in, const int* in_sizes, int n_in,
                              void* d_out, int out_size, void* d_ws, size_t ws_size,
                              hipStream_t stream) {
  const int*   enc_in   = (const int*)d_in[0];
  const int*   tgt_in   = (const int*)d_in[2];
  const float* emb      = (const float*)d_in[4];
  const float* enc_Wih  = (const float*)d_in[5];
  const float* enc_Whh  = (const float*)d_in[6];
  const float* enc_bih  = (const float*)d_in[7];
  const float* enc_bhh  = (const float*)d_in[8];
  const float* dec_Wih  = (const float*)d_in[9];
  const float* dec_Whh  = (const float*)d_in[10];
  const float* dec_bih  = (const float*)d_in[11];
  const float* dec_bhh  = (const float*)d_in[12];
  const float* att_Wenc = (const float*)d_in[13];
  const float* att_benc = (const float*)d_in[14];
  const float* att_Wdec = (const float*)d_in[15];
  const float* att_bdec = (const float*)d_in[16];
  const float* att_Wv   = (const float*)d_in[17];
  const float* att_bv   = (const float*)d_in[18];
  const float* out_W    = (const float*)d_in[19];
  const float* out_b    = (const float*)d_in[20];
  float* out = (float*)d_out;

  // workspace carve (256B aligned)
  char* ws = (char*)d_ws;
  size_t off = 0;
  auto carve = [&](size_t bytes) -> void* {
    off = (off + 255) & ~(size_t)255;
    void* p = ws + off;
    off += bytes;
    return p;
  };
  float* enc_emb = (float*)carve((size_t)MM * HH * 4);
  float* tgt_emb = (float*)carve((size_t)MM * HH * 4);
  float* enc_xp  = (float*)carve((size_t)MM * 3 * HH * 4);
  float* dec_xp  = (float*)carve((size_t)MM * 3 * HH * 4);
  float* enc_out = (float*)carve((size_t)MM * HH * 4);
  float* ge      = (float*)carve((size_t)MM * HH * 4);
  float* hbuf    = (float*)carve((size_t)BB * HH * 4);
  unsigned short* Abf = (unsigned short*)carve((size_t)MP * KP * 2);
  unsigned short* Wb  = (unsigned short*)carve((size_t)VP * KP * 2);

  // init pad region of A + convert out_W to bf16
  zero_u16<<<(MP * KP + 255) / 256, 256, 0, stream>>>(Abf, MP * KP);
  convert_w_bf16<<<(VP * KP + 255) / 256, 256, 0, stream>>>(out_W, Wb);

  // embeddings
  embed_kernel<<<(MM * HH + 255) / 256, 256, 0, stream>>>(enc_in, emb, enc_emb, MM);
  embed_kernel<<<(MM * HH + 255) / 256, 256, 0, stream>>>(tgt_in, emb, tgt_emb, MM);

  // input projections (batched over all timesteps)
  {
    dim3 blk(16, 16);
    dim3 g1((3 * HH + 15) / 16, (MM + 15) / 16);
    sgemm_nt_bias<<<g1, blk, 0, stream>>>(enc_emb, enc_Wih, enc_bih, enc_xp, MM, 3 * HH, HH);
    sgemm_nt_bias<<<g1, blk, 0, stream>>>(tgt_emb, dec_Wih, dec_bih, dec_xp, MM, 3 * HH, HH);
  }

  // fused encoder recurrence (per-batch-row blocks, no grid sync needed)
  encoder_kernel<<<BB, 256, 0, stream>>>(enc_xp, enc_Whh, enc_bhh, enc_out, hbuf);

  // loop-invariant attention keys: ge = enc_out @ att_Wenc^T + benc
  {
    dim3 blk(16, 16);
    dim3 g2((HH + 15) / 16, (MM + 15) / 16);
    sgemm_nt_bias<<<g2, blk, 0, stream>>>(enc_out, att_Wenc, att_benc, ge, MM, HH, HH);
  }

  // fused decoder recurrence + attention (h continues from encoder final state)
  decoder_kernel<<<BB, 256, 0, stream>>>(dec_xp, dec_Whh, dec_bhh,
                                         att_Wdec, att_bdec, att_Wv, att_bv,
                                         ge, enc_out, hbuf, Abf);

  // the 37 GFLOP vocab projection, deferred out of the scan: bf16 WMMA
  wmma_out_gemm<<<dim3(VP / 64, (MP / 16) / 8), 256, 0, stream>>>(Abf, Wb, out_b, out);
}
